// PolarAttention_84378927497567
// MI455X (gfx1250) — compile-verified, matrix operand order checked by reading, not run on hardware
//
#include <hip/hip_runtime.h>
#include <hip/hip_bf16.h>

#define DEVI __device__ __forceinline__

typedef __bf16 bf16_t;
typedef __attribute__((ext_vector_type(16))) __bf16 v16bf;
typedef __attribute__((ext_vector_type(8)))  float  v8f;

static constexpr int NTOK    = 262144;   // 2*32*64*64 tokens
static constexpr int TOKMASK = 131071;   // tokens per batch (32*64*64) - 1

struct U128 { unsigned int x[4]; };
union FragB16 {
  v16bf  v;
  U128   q[2];
  bf16_t h[16];
};
static_assert(sizeof(FragB16) == 32, "frag size");

// ---- WMMA helpers ------------------------------------------------------
// A-fragment (16x32 bf16, row-major source [row][stride]):
//   lanes 0-15: row M = lane, K = k0+{0..7} (q[0]) and k0+16..23 (q[1])
//   lanes 16-31: row M = lane-16, K = k0+8..15 and k0+24..31
// B-fragment uses the same loader on the TRANSPOSED weight [N][K] so that
// lane = N column with contiguous-K 16B loads (mirrors ISA 7.12.2 layout).
DEVI FragB16 load_frag(const bf16_t* base, int row0, int k0, int stride, int lane) {
  FragB16 f;
  const bf16_t* p = base + (size_t)(row0 + (lane & 15)) * stride + k0 + ((lane >> 4) * 8);
  __builtin_memcpy(&f.q[0], p, 16);
  __builtin_memcpy(&f.q[1], p + 16, 16);
  return f;
}

// Build an A-fragment from an f32 row-major tile (converts to bf16).
DEVI FragB16 cvt_frag(const float* base, int k0, int stride, int lane) {
  FragB16 f;
  const float* p0 = base + (size_t)(lane & 15) * stride + k0 + ((lane >> 4) * 8);
  const float* p1 = p0 + 16;
#pragma unroll
  for (int i = 0; i < 8; ++i) { f.h[i] = (bf16_t)p0[i]; f.h[8 + i] = (bf16_t)p1[i]; }
  return f;
}

DEVI v8f wmma_bf16(const FragB16& a, const FragB16& b, v8f c) {
  return __builtin_amdgcn_wmma_f32_16x16x32_bf16(false, a.v, false, b.v, (short)0, c,
                                                 false, false);
}

DEVI float wave_sum(float v) {
#pragma unroll
  for (int m = 16; m >= 1; m >>= 1) v += __shfl_xor(v, m, 32);
  return v;
}

// ---- Kernel 0: weight convert + transpose (fp32 [K][N] -> bf16 [N][K]) --
__global__ void wprep_kernel(const float* __restrict__ src, bf16_t* __restrict__ dst,
                             int K, int Nn) {
  int idx = blockIdx.x * blockDim.x + threadIdx.x;
  if (idx >= K * Nn) return;
  int k = idx / Nn, n = idx - k * Nn;
  dst[(size_t)n * K + k] = (bf16_t)src[idx];
}

// ---- Kernel 1: gather transpose + polar proj + LN1 ----------------------
__global__ __launch_bounds__(256) void prep_ln1_kernel(
    const float* __restrict__ x, const float* __restrict__ polar,
    const float* __restrict__ Wp, const float* __restrict__ bp,
    const float* __restrict__ g1, const float* __restrict__ b1,
    float* __restrict__ resid1, bf16_t* __restrict__ xnorm) {
  __shared__ __align__(16) float xt[64 * 132];   // 64 tokens x 128 ch, padded stride
  __shared__ float pt[6 * 64];
  __shared__ float swp[6 * 128];
  __shared__ float sbp[128];
  const int t  = threadIdx.x;
  const int T0 = blockIdx.x * 64;

  for (int i = t; i < 6 * 64; i += 256) {
    int p = i >> 6, tok = i & 63;
    int n = T0 + tok;
    pt[i] = polar[(size_t)((n >> 17) * 6 + p) * 131072 + (n & TOKMASK)];
  }
  for (int i = t; i < 6 * 128; i += 256) swp[i] = Wp[i];
  if (t < 128) sbp[t] = bp[t];
  __syncthreads();

  for (int i = t; i < 64 * 128; i += 256) {
    int c = i >> 6, tok = i & 63;
    int n = T0 + tok;
    float v = x[(size_t)((n >> 17) * 128 + c) * 131072 + (n & TOKMASK)];
    float acc = sbp[c];
#pragma unroll
    for (int p = 0; p < 6; ++p) acc += pt[p * 64 + tok] * swp[p * 128 + c];
    xt[tok * 132 + c] = v + acc;
  }
  __syncthreads();

  const int wave = t >> 5, lane = t & 31;
  float4 gg = *(const float4*)(g1 + lane * 4);
  float4 bb = *(const float4*)(b1 + lane * 4);
  for (int tk = wave * 8; tk < wave * 8 + 8; ++tk) {
    float4 v = *(const float4*)(&xt[tk * 132] + lane * 4);
    float mean = wave_sum(v.x + v.y + v.z + v.w) * (1.0f / 128.0f);
    float dx = v.x - mean, dy = v.y - mean, dz = v.z - mean, dw = v.w - mean;
    float var  = wave_sum(dx * dx + dy * dy + dz * dz + dw * dw) * (1.0f / 128.0f);
    float rstd = rsqrtf(var + 1e-5f);
    size_t o = (size_t)(T0 + tk) * 128 + lane * 4;
    *(float4*)(resid1 + o) = v;
    bf16_t pk[4];
    pk[0] = (bf16_t)(dx * rstd * gg.x + bb.x);
    pk[1] = (bf16_t)(dy * rstd * gg.y + bb.y);
    pk[2] = (bf16_t)(dz * rstd * gg.z + bb.z);
    pk[3] = (bf16_t)(dw * rstd * gg.w + bb.w);
    __builtin_memcpy(xnorm + o, pk, 8);
  }
}

// ---- Kernel 2: QKV GEMMs + cross-head attention + Wo + LN2 --------------
// LDS: 128KB transposed weights (shared) + 24KB/wave QKV scratch.
__global__ __launch_bounds__(128) void attn_kernel(
    const bf16_t* __restrict__ xnorm, const bf16_t* __restrict__ wAll, // wq|wk|wv|wo T
    const float* __restrict__ bq, const float* __restrict__ bk,
    const float* __restrict__ bv, const float* __restrict__ bo,
    const float* __restrict__ resid1,
    const float* __restrict__ g2, const float* __restrict__ b2,
    float* __restrict__ resid2, bf16_t* __restrict__ hnorm, int tilesPerWave) {
  extern __shared__ __align__(16) char smem[];
  bf16_t* sw = (bf16_t*)smem;                        // 4 x [128][128] bf16
  const int t = threadIdx.x, lane = t & 31, wave = t >> 5;

  {   // stage all four transposed weight matrices (contiguous in ws)
    uint4* d = (uint4*)sw;
    const uint4* s = (const uint4*)wAll;
    for (int i = t; i < 8192; i += 128) d[i] = s[i];
  }
  __syncthreads();
  const bf16_t* swq = sw;
  const bf16_t* swk = sw + 16384;
  const bf16_t* swv = sw + 32768;
  const bf16_t* swo = sw + 49152;
  float* qv = (float*)(smem + 131072 + wave * 24576);   // [16][128]
  float* kv = qv + 2048;
  float* vv = qv + 4096;

  float4 gg = *(const float4*)(g2 + lane * 4);
  float4 bb = *(const float4*)(b2 + lane * 4);

  const int gw = blockIdx.x * 4 + wave;
  for (int it = 0; it < tilesPerWave; ++it) {
    const int T = (gw * tilesPerWave + it) * 16;
    __builtin_prefetch(xnorm + (size_t)(T + 16) * 128, 0, 0);

    FragB16 af[4];
#pragma unroll
    for (int kc = 0; kc < 4; ++kc)
      af[kc] = load_frag(xnorm + (size_t)T * 128, 0, kc * 32, 128, lane);

    const bf16_t* wts[3]  = {swq, swk, swv};
    const float*  bias[3] = {bq, bk, bv};
    float*        dsts[3] = {qv, kv, vv};
#pragma unroll
    for (int mm = 0; mm < 3; ++mm) {
      // pairs of N-tiles: preload all B frags, two independent acc chains
      for (int nt = 0; nt < 8; nt += 2) {
        FragB16 b0[4], b1[4];
#pragma unroll
        for (int kc = 0; kc < 4; ++kc) {
          b0[kc] = load_frag(wts[mm], (nt + 0) * 16, kc * 32, 128, lane);
          b1[kc] = load_frag(wts[mm], (nt + 1) * 16, kc * 32, 128, lane);
        }
        v8f a0 = {}, a1 = {};
#pragma unroll
        for (int kc = 0; kc < 4; ++kc) {
          a0 = wmma_bf16(af[kc], b0[kc], a0);
          a1 = wmma_bf16(af[kc], b1[kc], a1);
        }
        int ch0 = (nt + 0) * 16 + (lane & 15);
        int ch1 = (nt + 1) * 16 + (lane & 15);
        float bs0 = bias[mm][ch0], bs1 = bias[mm][ch1];
#pragma unroll
        for (int r = 0; r < 8; ++r) {
          int tok = r + (lane >> 4) * 8;
          dsts[mm][tok * 128 + ch0] = a0[r] + bs0;   // D layout
          dsts[mm][tok * 128 + ch1] = a1[r] + bs1;
        }
      }
    }

    // per-token 8x8 cross-head attention; 32 lanes x 4 = 128 (tok,head) pairs.
    // attended overwrites Q rows of the same tok-group only (program order safe).
#pragma unroll
    for (int j = 0; j < 4; ++j) {
      int idx = j * 32 + lane;
      int tok = idx >> 3, head = idx & 7;
      float qr[16];
#pragma unroll
      for (int d = 0; d < 16; ++d) qr[d] = qv[tok * 128 + head * 16 + d];
      float s[8], mx = -1e30f;
#pragma unroll
      for (int g = 0; g < 8; ++g) {
        float a = 0.f;
#pragma unroll
        for (int d = 0; d < 16; ++d) a += qr[d] * kv[tok * 128 + g * 16 + d];
        s[g] = a * 0.25f;                               // 1/sqrt(16)
        mx = fmaxf(mx, s[g]);
      }
      float den = 0.f;
#pragma unroll
      for (int g = 0; g < 8; ++g) { s[g] = __expf(s[g] - mx); den += s[g]; }
      float inv = 1.0f / den;
#pragma unroll
      for (int d = 0; d < 16; ++d) {
        float a = 0.f;
#pragma unroll
        for (int g = 0; g < 8; ++g) a += s[g] * vv[tok * 128 + g * 16 + d];
        qv[tok * 128 + head * 16 + d] = a * inv;
      }
    }

    // out = attended @ Wo + bo + resid1  -> kv tile (f32)
    FragB16 aa[4];
#pragma unroll
    for (int kc = 0; kc < 4; ++kc) aa[kc] = cvt_frag(qv, kc * 32, 128, lane);
    for (int nt = 0; nt < 8; nt += 2) {
      FragB16 b0[4], b1[4];
#pragma unroll
      for (int kc = 0; kc < 4; ++kc) {
        b0[kc] = load_frag(swo, (nt + 0) * 16, kc * 32, 128, lane);
        b1[kc] = load_frag(swo, (nt + 1) * 16, kc * 32, 128, lane);
      }
      v8f a0 = {}, a1 = {};
#pragma unroll
      for (int kc = 0; kc < 4; ++kc) {
        a0 = wmma_bf16(aa[kc], b0[kc], a0);
        a1 = wmma_bf16(aa[kc], b1[kc], a1);
      }
      int ch0 = (nt + 0) * 16 + (lane & 15);
      int ch1 = (nt + 1) * 16 + (lane & 15);
      float bs0 = bo[ch0], bs1 = bo[ch1];
#pragma unroll
      for (int r = 0; r < 8; ++r) {
        int tok = r + (lane >> 4) * 8;
        kv[tok * 128 + ch0] = a0[r] + bs0 + resid1[(size_t)(T + tok) * 128 + ch0];
        kv[tok * 128 + ch1] = a1[r] + bs1 + resid1[(size_t)(T + tok) * 128 + ch1];
      }
    }

    // LN2: one wave per token iteration; write resid2 (f32) + hnorm (bf16)
    for (int tok = 0; tok < 16; ++tok) {
      float4 v = *(const float4*)(kv + tok * 128 + lane * 4);
      float mean = wave_sum(v.x + v.y + v.z + v.w) * (1.0f / 128.0f);
      float dx = v.x - mean, dy = v.y - mean, dz = v.z - mean, dw = v.w - mean;
      float var  = wave_sum(dx * dx + dy * dy + dz * dz + dw * dw) * (1.0f / 128.0f);
      float rstd = rsqrtf(var + 1e-5f);
      size_t o = (size_t)(T + tok) * 128 + lane * 4;
      *(float4*)(resid2 + o) = v;
      bf16_t pk[4];
      pk[0] = (bf16_t)(dx * rstd * gg.x + bb.x);
      pk[1] = (bf16_t)(dy * rstd * gg.y + bb.y);
      pk[2] = (bf16_t)(dz * rstd * gg.z + bb.z);
      pk[3] = (bf16_t)(dw * rstd * gg.w + bb.w);
      __builtin_memcpy(hnorm + o, pk, 8);
    }
  }
}

// ---- Kernel 3: MLP (128 -> 512 -> 128) + residual + scatter -------------
// LDS: W1T (128KB) + W2T (128KB) shared + 16KB/wave h-buffer = 288KB.
__global__ __launch_bounds__(64) void mlp_kernel(
    const bf16_t* __restrict__ hnorm,
    const bf16_t* __restrict__ w1T, const bf16_t* __restrict__ w2T,
    const float* __restrict__ bf1, const float* __restrict__ bf2,
    const float* __restrict__ resid2, float* __restrict__ out, int tilesPerWave) {
  extern __shared__ __align__(16) char smem[];
  bf16_t* s1 = (bf16_t*)smem;            // [512][128]
  bf16_t* s2 = s1 + 65536;               // [128][512]
  const int t = threadIdx.x, lane = t & 31, wave = t >> 5;
  bf16_t* hbuf = (bf16_t*)(smem + 262144 + wave * 16384);   // [16][512] bf16

  {
    uint4* d1 = (uint4*)s1; const uint4* r1 = (const uint4*)w1T;
    uint4* d2 = (uint4*)s2; const uint4* r2 = (const uint4*)w2T;
    for (int i = t; i < 8192; i += 64) { d1[i] = r1[i]; d2[i] = r2[i]; }
  }
  __syncthreads();

  const int gw = blockIdx.x * 2 + wave;
  for (int it = 0; it < tilesPerWave; ++it) {
    const int T = (gw * tilesPerWave + it) * 16;
    __builtin_prefetch(hnorm + (size_t)(T + 16) * 128, 0, 0);

    FragB16 af[4];
#pragma unroll
    for (int kc = 0; kc < 4; ++kc)
      af[kc] = load_frag(hnorm + (size_t)T * 128, 0, kc * 32, 128, lane);

    // FC1 + bias + ReLU -> bf16 LDS h-buffer (paired N-tiles, preloaded B)
    for (int nt = 0; nt < 32; nt += 2) {
      FragB16 b0[4], b1[4];
#pragma unroll
      for (int kc = 0; kc < 4; ++kc) {
        b0[kc] = load_frag(s1, (nt + 0) * 16, kc * 32, 128, lane);
        b1[kc] = load_frag(s1, (nt + 1) * 16, kc * 32, 128, lane);
      }
      v8f a0 = {}, a1 = {};
#pragma unroll
      for (int kc = 0; kc < 4; ++kc) {
        a0 = wmma_bf16(af[kc], b0[kc], a0);
        a1 = wmma_bf16(af[kc], b1[kc], a1);
      }
      int ch0 = (nt + 0) * 16 + (lane & 15);
      int ch1 = (nt + 1) * 16 + (lane & 15);
      float bs0 = bf1[ch0], bs1 = bf1[ch1];
#pragma unroll
      for (int r = 0; r < 8; ++r) {
        int tok = r + (lane >> 4) * 8;
        hbuf[tok * 512 + ch0] = (bf16_t)fmaxf(a0[r] + bs0, 0.f);
        hbuf[tok * 512 + ch1] = (bf16_t)fmaxf(a1[r] + bs1, 0.f);
      }
    }

    // FC2: preload all 16 A-fragments, paired N-tiles, K in two halves of 8
    FragB16 hf[16];
#pragma unroll
    for (int kc = 0; kc < 16; ++kc) hf[kc] = load_frag(hbuf, 0, kc * 32, 512, lane);
    v8f accs[8];
#pragma unroll
    for (int nt = 0; nt < 8; nt += 2) {
      v8f a0 = {}, a1 = {};
#pragma unroll
      for (int kh = 0; kh < 2; ++kh) {
        FragB16 b0[8], b1[8];
#pragma unroll
        for (int kc = 0; kc < 8; ++kc) {
          b0[kc] = load_frag(s2, (nt + 0) * 16, (kh * 8 + kc) * 32, 512, lane);
          b1[kc] = load_frag(s2, (nt + 1) * 16, (kh * 8 + kc) * 32, 512, lane);
        }
#pragma unroll
        for (int kc = 0; kc < 8; ++kc) {
          a0 = wmma_bf16(hf[kh * 8 + kc], b0[kc], a0);
          a1 = wmma_bf16(hf[kh * 8 + kc], b1[kc], a1);
        }
      }
      int ch0 = (nt + 0) * 16 + (lane & 15);
      int ch1 = (nt + 1) * 16 + (lane & 15);
      float bs0 = bf2[ch0], bs1 = bf2[ch1];
#pragma unroll
      for (int r = 0; r < 8; ++r) {
        int tok = r + (lane >> 4) * 8;
        a0[r] += bs0 + resid2[(size_t)(T + tok) * 128 + ch0];
        a1[r] += bs1 + resid2[(size_t)(T + tok) * 128 + ch1];
      }
      accs[nt] = a0;
      accs[nt + 1] = a1;
    }

    // LDS bounce (all hbuf reads are above in program order), then scatter
    float* obuf = (float*)hbuf;   // [16][128] f32
#pragma unroll
    for (int nt = 0; nt < 8; ++nt) {
      int ch = nt * 16 + (lane & 15);
#pragma unroll
      for (int r = 0; r < 8; ++r)
        obuf[(r + (lane >> 4) * 8) * 128 + ch] = accs[nt][r];
    }
    for (int i = 0; i < 64; ++i) {
      int idx = i * 32 + lane;
      int tok = idx & 15, c = idx >> 4;          // 16 token-contiguous stores per half-wave
      int n = T + tok;
      out[(size_t)((n >> 17) * 128 + c) * 131072 + (n & TOKMASK)] = obuf[tok * 128 + c];
    }
  }
}

// ---- Host launch --------------------------------------------------------
extern "C" void kernel_launch(void* const* d_in, const int* in_sizes, int n_in,
                              void* d_out, int out_size, void* d_ws, size_t ws_size,
                              hipStream_t stream) {
  (void)in_sizes; (void)n_in; (void)out_size; (void)ws_size;
  const float* x     = (const float*)d_in[0];
  const float* polar = (const float*)d_in[1];
  const float* Wq = (const float*)d_in[2];  const float* bq = (const float*)d_in[3];
  const float* Wk = (const float*)d_in[4];  const float* bk = (const float*)d_in[5];
  const float* Wv = (const float*)d_in[6];  const float* bv = (const float*)d_in[7];
  const float* Wp = (const float*)d_in[8];  const float* bp = (const float*)d_in[9];
  const float* Wo = (const float*)d_in[10]; const float* bo = (const float*)d_in[11];
  const float* g1 = (const float*)d_in[12]; const float* b1 = (const float*)d_in[13];
  const float* g2 = (const float*)d_in[14]; const float* b2 = (const float*)d_in[15];
  const float* W1 = (const float*)d_in[16]; const float* bf1 = (const float*)d_in[17];
  const float* W2 = (const float*)d_in[18]; const float* bf2 = (const float*)d_in[19];
  float* out = (float*)d_out;

  char* ws = (char*)d_ws;
  float*  resid1 = (float*)ws;                               // 128 MB
  bf16_t* xnorm  = (bf16_t*)(ws + 134217728);                //  64 MB
  float*  resid2 = (float*)(ws + 201326592);                 // 128 MB
  bf16_t* hnorm  = (bf16_t*)(ws + 335544320);                //  64 MB
  bf16_t* wqT    = (bf16_t*)(ws + 402653184);                // 4x32KB (contiguous)
  bf16_t* wkT = wqT + 16384;
  bf16_t* wvT = wkT + 16384;
  bf16_t* woT = wvT + 16384;
  bf16_t* w1T = woT + 16384;                                 // [512][128]
  bf16_t* w2T = w1T + 65536;                                 // [128][512]

  wprep_kernel<<<64, 256, 0, stream>>>(Wq, wqT, 128, 128);
  wprep_kernel<<<64, 256, 0, stream>>>(Wk, wkT, 128, 128);
  wprep_kernel<<<64, 256, 0, stream>>>(Wv, wvT, 128, 128);
  wprep_kernel<<<64, 256, 0, stream>>>(Wo, woT, 128, 128);
  wprep_kernel<<<256, 256, 0, stream>>>(W1, w1T, 128, 512);
  wprep_kernel<<<256, 256, 0, stream>>>(W2, w2T, 512, 128);

  prep_ln1_kernel<<<NTOK / 64, 256, 0, stream>>>(x, polar, Wp, bp, g1, b1,
                                                 resid1, xnorm);

  // 256 blocks x 4 waves = 1024 waves; 16384 tiles -> 16 tiles/wave
  (void)hipFuncSetAttribute((const void*)attn_kernel,
                            hipFuncAttributeMaxDynamicSharedMemorySize, 229376);
  attn_kernel<<<256, 128, 229376, stream>>>(xnorm, wqT, bq, bk, bv, bo, resid1,
                                            g2, b2, resid2, hnorm, 16);

  // 512 blocks x 2 waves = 1024 waves; 16 tiles/wave
  (void)hipFuncSetAttribute((const void*)mlp_kernel,
                            hipFuncAttributeMaxDynamicSharedMemorySize, 294912);
  mlp_kernel<<<512, 64, 294912, stream>>>(hnorm, w1T, w2T, bf1, bf2, resid2, out, 16);
}
